// CRF_50337016709511
// MI455X (gfx1250) — matrix operand order repaired and checked by reading, not compiled, not run on hardware
//
#include <hip/hip_runtime.h>

// ---------------------------------------------------------------------------
// CRF NLL on gfx1250:  out = forward_logZ_sum - gold_score_sum
// Chunked matrix-chain forward:  alpha_t = alpha_{t-1} (expT · diag(e_t)).
// Each (batch, chunk) wave computes Q = P_c^T = prod M_t^T via 32 WMMAs/step
// (64x64x64 bf16 GEMM), register-resident, ds_swizzle half-wave exchange.
// 4096 independent waves -> fills the machine; combine pass is tiny.
// ---------------------------------------------------------------------------

typedef __attribute__((ext_vector_type(16))) __bf16        v16bf;
typedef __attribute__((ext_vector_type(8)))  float         v8f;
typedef __attribute__((ext_vector_type(8)))  unsigned int  v8u;

static constexpr int Sdim   = 512;
static constexpr int Tdim   = 64;    // includes START = T-2, STOP = T-1
static constexpr int CHUNKS = 8;
static constexpr int CLEN   = 64;    // steps per chunk (last chunk runs 63)
static constexpr int WPITCH = 32 * 64;  // dwords per stored P^T (bf16 pairs)

__device__ __forceinline__ unsigned int pack_bf16(float lo, float hi) {
  unsigned short a = __builtin_bit_cast(unsigned short, (__bf16)lo);
  unsigned short b = __builtin_bit_cast(unsigned short, (__bf16)hi);
  return (unsigned int)a | ((unsigned int)b << 16);
}

template <int OFF>
__device__ __forceinline__ float swz_fmax(float m) {
  int o = __builtin_amdgcn_ds_swizzle(__builtin_bit_cast(int, m), OFF);
  return fmaxf(m, __builtin_bit_cast(float, o));
}
template <int OFF>
__device__ __forceinline__ int swz_iadd(int v) {
  return v + __builtin_amdgcn_ds_swizzle(v, OFF);
}

__global__ void crf_zero(float* out) { out[0] = 0.0f; }

// ------------------------------ gold score ---------------------------------
__global__ void __launch_bounds__(256) crf_gold(const float* __restrict__ feats,
                                                const int*   __restrict__ mask,
                                                const int*   __restrict__ tags,
                                                const float* __restrict__ trans,
                                                float* out) {
  const int b   = blockIdx.x;
  const int tid = threadIdx.x;
  __shared__ float red[256];
  __shared__ int   cred[256];

  float s   = 0.0f;
  int   cnt = 0;
  for (int t = tid; t < Sdim; t += 256) {
    if (mask[b * Sdim + t] != 0) {
      ++cnt;
      const int tg = tags[b * Sdim + t];
      const int pv = (t == 0) ? (Tdim - 2) : tags[b * Sdim + t - 1];  // START at t=0
      s += feats[((size_t)b * Sdim + t) * Tdim + tg] + trans[pv * Tdim + tg];
    }
  }
  red[tid] = s; cred[tid] = cnt;
  __syncthreads();
  for (int off = 128; off > 0; off >>= 1) {
    if (tid < off) { red[tid] += red[tid + off]; cred[tid] += cred[tid + off]; }
    __syncthreads();
  }
  if (tid == 0) {
    const int len   = cred[0];
    const int endId = tags[b * Sdim + (len - 1)];
    atomicAdd(out, -(red[0] + trans[endId * Tdim + (Tdim - 1)]));
  }
}

// --------------------------- chunk product ---------------------------------
// One wave per (b, c).  Q starts as I; per step Q <- (diag(e_t) expT^T) * Q.
// A-layout (16x32 bf16): lane lm -> M row; VGPR r -> K pair
//   k0 = 2r + hi*16 (within each 32-wide K block).
// B-layout (32x16 bf16): lanes 0-15 K=0..15, lanes 16-31 K=16..31, N = lm.
// C/D layout (f32 16x16): M = r + hi*8, N = lm.
__global__ void __launch_bounds__(32) crf_chunk(const float* __restrict__ feats,
                                                const int*   __restrict__ mask,
                                                const float* __restrict__ trans,
                                                unsigned int* __restrict__ wsW,
                                                float*        __restrict__ wsScale) {
  const int lane = threadIdx.x;
  const int hi   = lane >> 4;
  const int lm   = lane & 15;
  const int b    = blockIdx.x >> 3;
  const int c    = blockIdx.x & (CHUNKS - 1);

  // sequence length (wave-uniform): popcount of prefix mask via swizzle tree
  int len = 0;
  for (int i = 0; i < Sdim / 32; ++i) len += (mask[b * Sdim + i * 32 + lane] != 0);
  len = swz_iadd<0x041F>(len); len = swz_iadd<0x081F>(len);
  len = swz_iadd<0x101F>(len); len = swz_iadd<0x201F>(len); len = swz_iadd<0x401F>(len);

  // Resident A source: expT^T in A-fragment bf16 pairs.  (M=to, K=from)
  v16bf eTbf[4][2];
#pragma unroll
  for (int mt = 0; mt < 4; ++mt)
#pragma unroll
    for (int kf = 0; kf < 2; ++kf) {
      v8u w;
#pragma unroll
      for (int r = 0; r < 8; ++r) {
        const int from0 = kf * 32 + hi * 16 + 2 * r;
        const int to    = mt * 16 + lm;
        w[r] = pack_bf16(__expf(trans[from0 * Tdim + to]),
                         __expf(trans[(from0 + 1) * Tdim + to]));
      }
      eTbf[mt][kf] = __builtin_bit_cast(v16bf, w);
    }

  // Q = identity (bf16 B-layout)
  v8u bqu[2][4];
#pragma unroll
  for (int kb = 0; kb < 2; ++kb)
#pragma unroll
    for (int nt = 0; nt < 4; ++nt)
#pragma unroll
      for (int r = 0; r < 8; ++r) {
        const int K0 = kb * 32 + hi * 16 + 2 * r;
        const int N  = nt * 16 + lm;
        bqu[kb][nt][r] = (K0 == N ? 0x3F80u : 0u) | ((K0 + 1 == N) ? (0x3F80u << 16) : 0u);
      }

  float scaleAcc = 0.0f;
  const int t0 = 1 + c * CLEN;
  int tEnd = t0 + CLEN;
  if (tEnd > Sdim) tEnd = Sdim;
  if (tEnd > len)  tEnd = len;          // masked steps are identity -> stop early

  const v8f zeroC = {0.f, 0.f, 0.f, 0.f, 0.f, 0.f, 0.f, 0.f};

  for (int t = t0; t < tEnd; ++t) {
    if (t + 1 < tEnd)
      __builtin_prefetch(&feats[((size_t)b * Sdim + (t + 1)) * Tdim + lane * 2], 0, 0);

    // A = diag(e_t) * expT^T : scale resident fragments by per-row emission
    float e[4];
#pragma unroll
    for (int mt = 0; mt < 4; ++mt)
      e[mt] = __expf(feats[((size_t)b * Sdim + t) * Tdim + mt * 16 + lm]);

    v16bf Abf[4][2];
#pragma unroll
    for (int mt = 0; mt < 4; ++mt) {
      const __bf16 eb = (__bf16)e[mt];
      v16bf s;
#pragma unroll
      for (int i = 0; i < 16; ++i) s[i] = eb;
      Abf[mt][0] = eTbf[mt][0] * s;     // v_pk_mul_bf16
      Abf[mt][1] = eTbf[mt][1] * s;
    }

    // Q_new = A * Q : 32 x v_wmma_f32_16x16x32_bf16
    v8f d[4][4];
#pragma unroll
    for (int mt = 0; mt < 4; ++mt)
#pragma unroll
      for (int nt = 0; nt < 4; ++nt) {
        d[mt][nt] = __builtin_amdgcn_wmma_f32_16x16x32_bf16(
            false, Abf[mt][0], false, __builtin_bit_cast(v16bf, bqu[0][nt]),
            (short)0, zeroC, false, false);
        d[mt][nt] = __builtin_amdgcn_wmma_f32_16x16x32_bf16(
            false, Abf[mt][1], false, __builtin_bit_cast(v16bf, bqu[1][nt]),
            (short)0, d[mt][nt], false, false);
      }

    // global tile max (stability scale): local 128-max + 5 swizzle-xor hops
    float gm = 0.0f;
#pragma unroll
    for (int mt = 0; mt < 4; ++mt)
#pragma unroll
      for (int nt = 0; nt < 4; ++nt)
#pragma unroll
        for (int r = 0; r < 8; ++r) gm = fmaxf(gm, d[mt][nt][r]);
    gm = swz_fmax<0x041F>(gm); gm = swz_fmax<0x081F>(gm); gm = swz_fmax<0x101F>(gm);
    gm = swz_fmax<0x201F>(gm); gm = swz_fmax<0x401F>(gm);
    scaleAcc += __logf(gm);
    const float inv = 1.0f / gm;

    // repack D (rows hi*8+r) into B layout (K half hi*16): local bf16 pack,
    // then one SWAPX16 swizzle per dword to fetch the other half's rows.
    unsigned int pk[4][4][4];
#pragma unroll
    for (int mt = 0; mt < 4; ++mt)
#pragma unroll
      for (int nt = 0; nt < 4; ++nt)
#pragma unroll
        for (int q = 0; q < 4; ++q)
          pk[mt][nt][q] = pack_bf16(d[mt][nt][2 * q] * inv, d[mt][nt][2 * q + 1] * inv);

#pragma unroll
    for (int kb = 0; kb < 2; ++kb)
#pragma unroll
      for (int nt = 0; nt < 4; ++nt)
#pragma unroll
        for (int q = 0; q < 4; ++q) {
          const unsigned int pA = pk[2 * kb][nt][q];       // tile 2kb+0 (my half rows)
          const unsigned int pB = pk[2 * kb + 1][nt][q];   // tile 2kb+1
          const unsigned int sendv = hi ? pA : pB;         // what my partner needs
          const unsigned int recv  = (unsigned int)__builtin_amdgcn_ds_swizzle(
              (int)sendv, 0x401F);                         // SWAPX16
          const unsigned int loc   = hi ? pB : pA;         // my needed tile, my rows
          bqu[kb][nt][q]     = hi ? recv : loc;            // rows 0..7  (half 0)
          bqu[kb][nt][q + 4] = hi ? loc  : recv;           // rows 8..15 (half 1)
        }
  }

  // store P^T (bf16 pairs): word (iPair, j) with row i = 2*iPair + p
  unsigned int* W = wsW + (size_t)(b * CHUNKS + c) * WPITCH;
#pragma unroll
  for (int kb = 0; kb < 2; ++kb)
#pragma unroll
    for (int nt = 0; nt < 4; ++nt)
#pragma unroll
      for (int r = 0; r < 8; ++r)
        W[(kb * 16 + hi * 8 + r) * 64 + nt * 16 + lm] = bqu[kb][nt][r];
  if (lane == 0) wsScale[b * CHUNKS + c] = scaleAcc;
}

// ------------------------------ combine ------------------------------------
// Per batch: alpha = init; for c: alpha <- alpha * P_c (read as Q=P^T rows),
// renormalize, accumulate scales; then STOP reduction.
__global__ void __launch_bounds__(64) crf_combine(const float* __restrict__ feats,
                                                  const float* __restrict__ trans,
                                                  const unsigned int* __restrict__ wsW,
                                                  const float* __restrict__ wsScale,
                                                  float* out) {
  const int b  = blockIdx.x;
  const int to = threadIdx.x;          // 0..63
  __shared__ float alphaSh[64];
  __shared__ float red[64];

  // init: feats[b,0,:] + trans[START,:]
  float x = feats[(size_t)b * Sdim * Tdim + to] + trans[(Tdim - 2) * Tdim + to];
  red[to] = x;
  __syncthreads();
  for (int off = 32; off > 0; off >>= 1) {
    if (to < off) red[to] = fmaxf(red[to], red[to + off]);
    __syncthreads();
  }
  float cLog = red[0];
  __syncthreads();
  alphaSh[to] = __expf(x - cLog);
  __syncthreads();

  for (int c = 0; c < CHUNKS; ++c) {
    const unsigned int* W = wsW + (size_t)(b * CHUNKS + c) * WPITCH + (to >> 1) * 64;
    const bool hiHalf = (to & 1);
    float s = 0.0f;
#pragma unroll 8
    for (int j = 0; j < 64; ++j) {
      const unsigned int u = W[j];
      const float q = __builtin_bit_cast(float, hiHalf ? (u & 0xFFFF0000u) : (u << 16));
      s += alphaSh[j] * q;
    }
    __syncthreads();
    red[to] = s;
    __syncthreads();
    for (int off = 32; off > 0; off >>= 1) {
      if (to < off) red[to] = fmaxf(red[to], red[to + off]);
      __syncthreads();
    }
    const float m = red[0];
    __syncthreads();
    alphaSh[to] = s / m;
    cLog += __logf(m) + wsScale[b * CHUNKS + c];
    __syncthreads();
  }

  // final transition to STOP
  red[to] = alphaSh[to] * __expf(trans[to * Tdim + (Tdim - 1)]);
  __syncthreads();
  for (int off = 32; off > 0; off >>= 1) {
    if (to < off) red[to] += red[to + off];
    __syncthreads();
  }
  if (to == 0) atomicAdd(out, cLog + __logf(red[0]));
}

// ---------------------------------------------------------------------------
extern "C" void kernel_launch(void* const* d_in, const int* in_sizes, int n_in,
                              void* d_out, int out_size, void* d_ws, size_t ws_size,
                              hipStream_t stream) {
  const float* feats = (const float*)d_in[0];   // (512,512,64) f32
  const int*   mask  = (const int*)d_in[1];     // (512,512) bool->int
  const int*   tags  = (const int*)d_in[2];     // (512,512) int32
  const float* trans = (const float*)d_in[3];   // (64,64) f32
  float* out = (float*)d_out;                   // scalar

  const int B = in_sizes[1] / Sdim;             // 512

  unsigned int* wsW     = (unsigned int*)d_ws;                    // 32 MB of P^T
  float*        wsScale = (float*)((char*)d_ws + (size_t)B * CHUNKS * WPITCH * 4);

  crf_zero<<<1, 1, 0, stream>>>(out);
  crf_gold<<<B, 256, 0, stream>>>(feats, mask, tags, trans, out);
  crf_chunk<<<B * CHUNKS, 32, 0, stream>>>(feats, mask, trans, wsW, wsScale);
  crf_combine<<<B, 64, 0, stream>>>(feats, trans, wsW, wsScale, out);
}
